// DecoderCacheLayer_25451976196640
// MI455X (gfx1250) — matrix operand order repaired and checked by reading, not compiled, not run on hardware
//
#include <hip/hip_runtime.h>
#include <hip/hip_bf16.h>
#include <math.h>

// ---------------- types ----------------
typedef __bf16 bf16_t;
typedef __attribute__((ext_vector_type(16))) __bf16 v16bf;
typedef __attribute__((ext_vector_type(8)))  float  v8f;

union FragBF {
    v16bf v;
    float4 q[2];   // v16bf = 32 bytes = 2x float4
};

static __device__ __forceinline__ bf16_t f2bf(float f) {
    unsigned u = __builtin_bit_cast(unsigned, f);
    unsigned r = u + 0x7fffu + ((u >> 16) & 1u);   // round-to-nearest-even
    unsigned short h = (unsigned short)(r >> 16);
    return __builtin_bit_cast(bf16_t, h);
}

static __device__ __forceinline__ float sigmoidf_(float x) {
    return 1.0f / (1.0f + expf(-x));
}

static __device__ __forceinline__ float gelu_tanh(float x) {
    const float k = 0.7978845608028654f;
    float inner = k * (x + 0.044715f * x * x * x);
    return 0.5f * x * (1.0f + tanhf(inner));
}

// ---------------- WMMA fragment loads ----------------
// A-matrix 16x32 bf16 layout (wave32):
// lanes 0-15 row M = lane, halves 0..7 -> K kb+0..7, halves 8..15 -> K kb+16..23
// lanes 16-31 same rows, halves 0..7 -> K kb+8..15, halves 8..15 -> K kb+24..31
static __device__ __forceinline__ v16bf load_frag_a(const bf16_t* __restrict__ A,
                                                    int mrow, int lda, int kb, int lane) {
    int m  = mrow + (lane & 15);
    int hi = lane >> 4;
    const bf16_t* p = A + (size_t)m * lda + kb + hi * 8;
    FragBF fr;
    fr.q[0] = *(const float4*)(p);
    fr.q[1] = *(const float4*)(p + 16);
    return fr.v;
}

// B-matrix 32x16 bf16, W pre-transposed as Wt[N][K]:
// lane column N = lane&15, lanes 0-15 halves 0..7 -> K kb+0..7, 8..15 -> K kb+16..23; lanes 16-31 halves 0..7 -> K kb+8..15, 8..15 -> K kb+24..31
static __device__ __forceinline__ v16bf load_frag_b(const bf16_t* __restrict__ Wt,
                                                    int nb, int ldb, int kb, int lane) {
    int n   = nb + (lane & 15);
    int khi = (lane >> 4) * 8;
    const bf16_t* p = Wt + (size_t)n * ldb + kb + khi;
    FragBF fr;
    fr.q[0] = *(const float4*)(p);
    fr.q[1] = *(const float4*)(p + 16);
    return fr.v;
}

#define WMMA_BF16(A_, B_, C_) \
    __builtin_amdgcn_wmma_f32_16x16x32_bf16(false, (A_), false, (B_), (short)0, (C_), false, false)

// ---------------- kernels ----------------
// float -> bf16 transposed weight prep: Wt[z][n][k] = W[z][k][n]
__global__ void k_transpose(const float* __restrict__ W, bf16_t* __restrict__ Wt, int K, int N) {
    int n = blockIdx.x * 16 + threadIdx.x;
    int k = blockIdx.y * 16 + threadIdx.y;
    size_t off = (size_t)blockIdx.z * K * N;
    Wt[off + (size_t)n * K + k] = f2bf(W[off + (size_t)k * N + n]);
}

__global__ void k_f2bf(const float* __restrict__ in, bf16_t* __restrict__ out, size_t n) {
    size_t i = (size_t)blockIdx.x * blockDim.x + threadIdx.x;
    if (i < n) out[i] = f2bf(in[i]);
}

// C[M,N] = A[M,K]bf16 @ Wt[N,K]bf16, fp32 accumulate via v_wmma_f32_16x16x32_bf16.
// block = 256 thr = 8 waves arranged 4(M) x 2(N); wave tile 32(M) x 64(N);
// block tile 128(M) x 128(N); grid (N/128, M/128).
__global__ __launch_bounds__(256) void k_gemm(const bf16_t* __restrict__ A,
                                              const bf16_t* __restrict__ Wt,
                                              float* __restrict__ C, int K, int N) {
    int lane = threadIdx.x & 31;
    int wave = threadIdx.x >> 5;
    int mrow = blockIdx.y * 128 + (wave & 3) * 32;
    int nb0  = blockIdx.x * 128 + (wave >> 2) * 64;
    v8f c00 = {}, c01 = {}, c10 = {}, c11 = {};
    v8f c20 = {}, c21 = {}, c30 = {}, c31 = {};
    for (int kb = 0; kb < K; kb += 32) {
        v16bf a0 = load_frag_a(A, mrow,      K, kb, lane);
        v16bf a1 = load_frag_a(A, mrow + 16, K, kb, lane);
        v16bf b0 = load_frag_b(Wt, nb0,      K, kb, lane);
        v16bf b1 = load_frag_b(Wt, nb0 + 16, K, kb, lane);
        v16bf b2 = load_frag_b(Wt, nb0 + 32, K, kb, lane);
        v16bf b3 = load_frag_b(Wt, nb0 + 48, K, kb, lane);
        c00 = WMMA_BF16(a0, b0, c00);  c01 = WMMA_BF16(a1, b0, c01);
        c10 = WMMA_BF16(a0, b1, c10);  c11 = WMMA_BF16(a1, b1, c11);
        c20 = WMMA_BF16(a0, b2, c20);  c21 = WMMA_BF16(a1, b2, c21);
        c30 = WMMA_BF16(a0, b3, c30);  c31 = WMMA_BF16(a1, b3, c31);
    }
    int mc = mrow + (lane >> 4) * 8;
    int nc = lane & 15;
    for (int r = 0; r < 8; ++r) {
        size_t ro0 = (size_t)(mc + r) * N;
        size_t ro1 = (size_t)(mc + 16 + r) * N;
        C[ro0 + nb0 + nc]      = c00[r];  C[ro1 + nb0 + nc]      = c01[r];
        C[ro0 + nb0 + 16 + nc] = c10[r];  C[ro1 + nb0 + 16 + nc] = c11[r];
        C[ro0 + nb0 + 32 + nc] = c20[r];  C[ro1 + nb0 + 32 + nc] = c21[r];
        C[ro0 + nb0 + 48 + nc] = c30[r];  C[ro1 + nb0 + 48 + nc] = c31[r];
    }
}

// Causal dilated conv as 5 accumulated WMMA GEMMs with per-tap shifted A rows.
// h[m,n] += gelu( sum_t LN[m - (4-t)*dil, :] @ Wt5[t][n][:] + bias[n] )
// Same 32x64 wave tile / 128x128 block tile as k_gemm; grid (D/128, M/128).
__global__ __launch_bounds__(256) void k_conv(const bf16_t* __restrict__ Aln,
                                              const bf16_t* __restrict__ Wt5,
                                              const float* __restrict__ bias,
                                              float* __restrict__ h, int dil) {
    const int D = 1024;
    int lane = threadIdx.x & 31;
    int wave = threadIdx.x >> 5;
    int mrow = blockIdx.y * 128 + (wave & 3) * 32;
    int nb0  = blockIdx.x * 128 + (wave >> 2) * 64;
    int hi   = lane >> 4;
    int m0 = mrow + (lane & 15);
    int m1 = m0 + 16;
    int b0i = m0 >> 12, s0 = m0 & 4095;
    int b1i = m1 >> 12, s1 = m1 & 4095;
    v8f c00 = {}, c01 = {}, c10 = {}, c11 = {};
    v8f c20 = {}, c21 = {}, c30 = {}, c31 = {};
    for (int t = 0; t < 5; ++t) {
        int sh = (4 - t) * dil;
        int sr0 = s0 - sh;  bool v0 = sr0 >= 0;
        int sr1 = s1 - sh;  bool v1 = sr1 >= 0;
        size_t ar0 = (size_t)(b0i << 12) + (v0 ? sr0 : 0);
        size_t ar1 = (size_t)(b1i << 12) + (v1 ? sr1 : 0);
        const bf16_t* Wt = Wt5 + (size_t)t * D * D;
        for (int kb = 0; kb < D; kb += 32) {
            const bf16_t* p0 = Aln + ar0 * D + kb + hi * 8;
            const bf16_t* p1 = Aln + ar1 * D + kb + hi * 8;
            FragBF fa0, fa1;
            fa0.q[0] = *(const float4*)(p0);
            fa0.q[1] = *(const float4*)(p0 + 16);
            fa1.q[0] = *(const float4*)(p1);
            fa1.q[1] = *(const float4*)(p1 + 16);
            if (!v0) { fa0.q[0] = make_float4(0.f,0.f,0.f,0.f); fa0.q[1] = make_float4(0.f,0.f,0.f,0.f); }
            if (!v1) { fa1.q[0] = make_float4(0.f,0.f,0.f,0.f); fa1.q[1] = make_float4(0.f,0.f,0.f,0.f); }
            v16bf a0 = fa0.v;
            v16bf a1 = fa1.v;
            v16bf b0 = load_frag_b(Wt, nb0,      D, kb, lane);
            v16bf b1 = load_frag_b(Wt, nb0 + 16, D, kb, lane);
            v16bf b2 = load_frag_b(Wt, nb0 + 32, D, kb, lane);
            v16bf b3 = load_frag_b(Wt, nb0 + 48, D, kb, lane);
            c00 = WMMA_BF16(a0, b0, c00);  c01 = WMMA_BF16(a1, b0, c01);
            c10 = WMMA_BF16(a0, b1, c10);  c11 = WMMA_BF16(a1, b1, c11);
            c20 = WMMA_BF16(a0, b2, c20);  c21 = WMMA_BF16(a1, b2, c21);
            c30 = WMMA_BF16(a0, b3, c30);  c31 = WMMA_BF16(a1, b3, c31);
        }
    }
    int mc = mrow + hi * 8;
    int nc = lane & 15;
    for (int r = 0; r < 8; ++r) {
        size_t ro0 = (size_t)(mc + r) * D;
        size_t ro1 = (size_t)(mc + 16 + r) * D;
        float v0s[4] = { c00[r], c10[r], c20[r], c30[r] };
        float v1s[4] = { c01[r], c11[r], c21[r], c31[r] };
        for (int t2 = 0; t2 < 4; ++t2) {
            int n = nb0 + t2 * 16 + nc;
            float y0 = v0s[t2] + bias[n];
            float y1 = v1s[t2] + bias[n];
            h[ro0 + n] = h[ro0 + n] + gelu_tanh(y0);
            h[ro1 + n] = h[ro1 + n] + gelu_tanh(y1);
        }
    }
}

// g[m] = sigmoid( dot(X[m,:], w) )
__global__ __launch_bounds__(256) void k_rowgate(const float* __restrict__ X,
                                                 const float* __restrict__ w,
                                                 float* __restrict__ g, int D) {
    __shared__ float red[256];
    size_t m = blockIdx.x;
    int tid = threadIdx.x;
    float acc = 0.f;
    const float* row = X + m * D;
    for (int d = tid; d < D; d += 256) acc += row[d] * w[d];
    red[tid] = acc;
    __syncthreads();
    for (int st = 128; st > 0; st >>= 1) {
        if (tid < st) red[tid] += red[tid + st];
        __syncthreads();
    }
    if (tid == 0) g[m] = sigmoidf_(red[0]);
}

// Y = X + g[row]*T
__global__ void k_addgate(const float* __restrict__ X, const float* __restrict__ g,
                          const float* __restrict__ T, float* __restrict__ Y,
                          size_t n, int D) {
    size_t i = (size_t)blockIdx.x * blockDim.x + threadIdx.x;
    if (i < n) Y[i] = X[i] + g[i / (size_t)D] * T[i];
}

// LayerNorm per row; write bf16 (for next GEMM) and optionally fp32
__global__ __launch_bounds__(256) void k_ln(const float* __restrict__ H,
                                            const float* __restrict__ sP,
                                            const float* __restrict__ bP,
                                            bf16_t* __restrict__ obf,
                                            float* __restrict__ of32) {
    const int D = 1024;
    __shared__ float rs[256], rq[256];
    size_t m = blockIdx.x;
    int tid = threadIdx.x;
    const float* row = H + m * D;
    float s1 = 0.f, s2 = 0.f;
    for (int d = tid; d < D; d += 256) { float v = row[d]; s1 += v; s2 += v * v; }
    rs[tid] = s1; rq[tid] = s2;
    __syncthreads();
    for (int st = 128; st > 0; st >>= 1) {
        if (tid < st) { rs[tid] += rs[tid + st]; rq[tid] += rq[tid + st]; }
        __syncthreads();
    }
    float mean = rs[0] * (1.f / D);
    float var  = rq[0] * (1.f / D) - mean * mean;
    float inv  = rsqrtf(var + 1e-5f);
    for (int d = tid; d < D; d += 256) {
        float v = (row[d] - mean) * inv * sP[d] + bP[d];
        obf[m * D + d] = f2bf(v);
        if (of32) of32[m * D + d] = v;
    }
}

// LTM attention read: R[m,:] = softmax(Q[m,:]@cache_b^T * scale) @ cache_b
__global__ __launch_bounds__(64) void k_attn_read(const float* __restrict__ Q,
                                                  const float* __restrict__ cache,
                                                  float* __restrict__ R, float scale) {
    const int DC = 256, KK = 64;
    __shared__ float sm[64];
    size_t m = blockIdx.x;
    int b = (int)(m >> 12);
    int t = threadIdx.x;
    const float* cb = cache + (size_t)b * KK * DC;
    const float* q  = Q + m * DC;
    const float* ck = cb + (size_t)t * DC;
    float dot = 0.f;
    for (int c = 0; c < DC; ++c) dot += q[c] * ck[c];
    float my = dot * scale;
    sm[t] = my;
    __syncthreads();
    float mx = -1e30f;
    for (int j = 0; j < KK; ++j) mx = fmaxf(mx, sm[j]);
    float sum = 0.f;
    for (int j = 0; j < KK; ++j) sum += expf(sm[j] - mx);
    __syncthreads();
    sm[t] = expf(my - mx) / sum;
    __syncthreads();
    int c0 = t * 4;
    float r0 = 0, r1 = 0, r2 = 0, r3 = 0;
    for (int j = 0; j < KK; ++j) {
        float aj = sm[j];
        const float* cj = cb + (size_t)j * DC + c0;
        r0 += aj * cj[0]; r1 += aj * cj[1]; r2 += aj * cj[2]; r3 += aj * cj[3];
    }
    R[m * DC + c0] = r0; R[m * DC + c0 + 1] = r1;
    R[m * DC + c0 + 2] = r2; R[m * DC + c0 + 3] = r3;
}

// WM attention read (8 keys, softmax then *valid, no renorm)
__global__ __launch_bounds__(64) void k_attn_wm(const float* __restrict__ Q,
                                                const float* __restrict__ wm,
                                                float* __restrict__ R) {
    const int DC = 256, NK = 8, STR = 257;
    __shared__ float aw[8];
    size_t m = blockIdx.x;
    int b = (int)(m >> 12);
    int t = threadIdx.x;
    const float* wb = wm + (size_t)b * NK * STR;
    if (t < NK) {
        float dot = 0.f;
        const float* ck = wb + (size_t)t * STR;
        const float* q  = Q + m * DC;
        for (int c = 0; c < DC; ++c) dot += q[c] * ck[c];
        aw[t] = dot * 0.0625f;
    }
    __syncthreads();
    float mx = -1e30f;
    for (int j = 0; j < NK; ++j) mx = fmaxf(mx, aw[j]);
    float sum = 0.f;
    for (int j = 0; j < NK; ++j) sum += expf(aw[j] - mx);
    __syncthreads();
    if (t < NK) aw[t] = (expf(aw[t] - mx) / sum) * wb[(size_t)t * STR + 256];
    __syncthreads();
    int c0 = t * 4;
    float r0 = 0, r1 = 0, r2 = 0, r3 = 0;
    for (int j = 0; j < NK; ++j) {
        float aj = aw[j];
        const float* cj = wb + (size_t)j * STR + c0;
        r0 += aj * cj[0]; r1 += aj * cj[1]; r2 += aj * cj[2]; r3 += aj * cj[3];
    }
    R[m * DC + c0] = r0; R[m * DC + c0 + 1] = r1;
    R[m * DC + c0 + 2] = r2; R[m * DC + c0 + 3] = r3;
}

// wl[m,k] = softmax_k(Q[m]@cache_b^T * scale) * gate[m]
__global__ __launch_bounds__(64) void k_attn_w(const float* __restrict__ Q,
                                               const float* __restrict__ cache,
                                               const float* __restrict__ gate,
                                               float* __restrict__ Wout, float scale) {
    const int DC = 256, KK = 64;
    __shared__ float sm[64];
    size_t m = blockIdx.x;
    int b = (int)(m >> 12);
    int t = threadIdx.x;
    const float* cb = cache + (size_t)b * KK * DC;
    const float* q  = Q + m * DC;
    const float* ck = cb + (size_t)t * DC;
    float dot = 0.f;
    for (int c = 0; c < DC; ++c) dot += q[c] * ck[c];
    float my = dot * scale;
    sm[t] = my;
    __syncthreads();
    float mx = -1e30f;
    for (int j = 0; j < KK; ++j) mx = fmaxf(mx, sm[j]);
    float sum = 0.f;
    for (int j = 0; j < KK; ++j) sum += expf(sm[j] - mx);
    Wout[m * KK + t] = (expf(my - mx) / sum) * gate[m];
}

// wv[m,k] = softmax_k(out[m]@Ws) * g[m]   (KW=8)
__global__ __launch_bounds__(64) void k_wv(const float* __restrict__ O,
                                           const float* __restrict__ Ws,
                                           const float* __restrict__ g,
                                           float* __restrict__ wv) {
    const int D = 1024, NK = 8;
    __shared__ float sw[8];
    size_t m = blockIdx.x;
    int t = threadIdx.x;
    if (t < NK) {
        float dot = 0.f;
        const float* row = O + m * D;
        for (int d = 0; d < D; ++d) dot += row[d] * Ws[(size_t)d * NK + t];
        sw[t] = dot;
    }
    __syncthreads();
    if (t < NK) {
        float mx = -1e30f;
        for (int j = 0; j < NK; ++j) mx = fmaxf(mx, sw[j]);
        float sum = 0.f;
        for (int j = 0; j < NK; ++j) sum += expf(sw[j] - mx);
        wv[m * NK + t] = (expf(sw[t] - mx) / sum) * g[m];
    }
}

// working-memory update: blocks = B*KW, threads = DC
__global__ __launch_bounds__(256) void k_upd_wm(const float* __restrict__ wv,
                                                const float* __restrict__ vw,
                                                const float* __restrict__ wmin,
                                                float* __restrict__ owm) {
    const int S = 4096, DC = 256, NK = 8, STR = 257;
    int blk = blockIdx.x;
    int b = blk >> 3, k = blk & 7;
    int c = threadIdx.x;
    size_t base = (size_t)b * S;
    float num = 0.f, den = 0.f;
    for (int s = 0; s < S; ++s) {
        size_t m = base + s;
        float w = wv[m * NK + k];
        num += w * vw[m * DC + c];
        den += w;
    }
    float avg = num / (den + 1e-6f);
    float alpha = tanhf(den);
    size_t wi = ((size_t)b * NK + k) * STR;
    owm[wi + c] = (1.f - alpha) * wmin[wi + c] + alpha * avg;
    if (c == 0) {
        float valid = wmin[wi + 256];
        owm[wi + 256] = valid + alpha * (1.f - valid);
    }
}

// LTM cache update: blocks = B*K, threads = DC
__global__ __launch_bounds__(256) void k_upd_cache(const float* __restrict__ wl,
                                                   const float* __restrict__ vl,
                                                   const float* __restrict__ cin,
                                                   float* __restrict__ oc) {
    const int S = 4096, DC = 256, KK = 64;
    int blk = blockIdx.x;
    int b = blk >> 6, k = blk & 63;
    int c = threadIdx.x;
    size_t base = (size_t)b * S;
    float num = 0.f, den = 0.f;
    for (int s = 0; s < S; ++s) {
        size_t m = base + s;
        float w = wl[m * KK + k];
        num += w * vl[m * DC + c];
        den += w;
    }
    float avg = num / (den + 1e-6f);
    float al  = tanhf(den);
    size_t ci = ((size_t)b * KK + k) * DC;
    oc[ci + c] = (1.f - al) * cin[ci + c] + al * avg;
}

// ---------------- host launch ----------------
extern "C" void kernel_launch(void* const* d_in, const int* in_sizes, int n_in,
                              void* d_out, int out_size, void* d_ws, size_t ws_size,
                              hipStream_t stream) {
    const int Bb = 4, S = 4096, D = 1024, DC = 256, KK = 64, NK = 8;
    const size_t M = (size_t)Bb * S;   // 16384

    const float* x        = (const float*)d_in[0];
    const float* cache    = (const float*)d_in[1];
    const float* wm       = (const float*)d_in[2];
    const float* Wq_ltm   = (const float*)d_in[3];
    const float* Wo_ltm   = (const float*)d_in[4];
    const float* wgr_ltm  = (const float*)d_in[5];
    const float* Wv_ltm   = (const float*)d_in[6];
    const float* wgw_ltm  = (const float*)d_in[7];
    const float* Wq_wm    = (const float*)d_in[8];
    const float* Wo_wm    = (const float*)d_in[9];
    const float* wgr_wm   = (const float*)d_in[10];
    const float* Wv_wm    = (const float*)d_in[11];
    const float* Ws_wm    = (const float*)d_in[12];
    const float* wgw_wm   = (const float*)d_in[13];
    const float* conv_w0  = (const float*)d_in[14];
    const float* conv_b0  = (const float*)d_in[15];
    const float* ln_s0    = (const float*)d_in[16];
    const float* ln_b0    = (const float*)d_in[17];
    const float* conv_w1  = (const float*)d_in[18];
    const float* conv_b1  = (const float*)d_in[19];
    const float* ln_s1    = (const float*)d_in[20];
    const float* ln_b1    = (const float*)d_in[21];
    const float* post_s   = (const float*)d_in[22];
    const float* post_b   = (const float*)d_in[23];

    float* out    = (float*)d_out;                         // [M, D]
    float* ocache = out + M * D;                           // [B, 64, 256]
    float* owm    = ocache + (size_t)Bb * KK * DC;         // [B, 8, 257]

    // workspace carve-up
    char*  ws  = (char*)d_ws;
    size_t off = 0;
    auto carve = [&](size_t bytes) {
        void* p = ws + off;
        off = (off + bytes + 255) & ~(size_t)255;
        return p;
    };
    bf16_t* wq_ltm_t = (bf16_t*)carve((size_t)DC * D * 2);
    bf16_t* wo_ltm_t = (bf16_t*)carve((size_t)D * DC * 2);
    bf16_t* wq_wm_t  = (bf16_t*)carve((size_t)DC * D * 2);
    bf16_t* wo_wm_t  = (bf16_t*)carve((size_t)D * DC * 2);
    bf16_t* wv_wm_t  = (bf16_t*)carve((size_t)DC * D * 2);
    bf16_t* wv_ltm_t = (bf16_t*)carve((size_t)DC * D * 2);
    bf16_t* convt0   = (bf16_t*)carve((size_t)5 * D * D * 2);
    bf16_t* convt1   = (bf16_t*)carve((size_t)5 * D * D * 2);
    bf16_t* ab0      = (bf16_t*)carve(M * D * 2);
    bf16_t* ab1      = (bf16_t*)carve(M * DC * 2);
    float*  f0       = (float*)carve(M * D * 4);
    float*  f1       = (float*)carve(M * D * 4);
    float*  f2       = (float*)carve(M * DC * 4);
    float*  f3       = (float*)carve(M * DC * 4);
    float*  wvb      = (float*)carve(M * NK * 4);
    float*  wlb      = (float*)carve(M * KK * 4);
    float*  gA       = (float*)carve(M * 4);
    float*  gB       = (float*)carve(M * 4);
    (void)ws_size; (void)in_sizes; (void)n_in; (void)out_size;

    const dim3 tb16(16, 16);
    const float sc = 0.0625f;   // 1/sqrt(DC), TEMP=1

    // weight prep (bf16 + transpose to [N][K])
    k_transpose<<<dim3(DC / 16, D / 16, 1), tb16, 0, stream>>>(Wq_ltm, wq_ltm_t, D, DC);
    k_transpose<<<dim3(D / 16, DC / 16, 1), tb16, 0, stream>>>(Wo_ltm, wo_ltm_t, DC, D);
    k_transpose<<<dim3(DC / 16, D / 16, 1), tb16, 0, stream>>>(Wq_wm, wq_wm_t, D, DC);
    k_transpose<<<dim3(D / 16, DC / 16, 1), tb16, 0, stream>>>(Wo_wm, wo_wm_t, DC, D);
    k_transpose<<<dim3(DC / 16, D / 16, 1), tb16, 0, stream>>>(Wv_wm, wv_wm_t, D, DC);
    k_transpose<<<dim3(DC / 16, D / 16, 1), tb16, 0, stream>>>(Wv_ltm, wv_ltm_t, D, DC);
    k_transpose<<<dim3(D / 16, D / 16, 5), tb16, 0, stream>>>(conv_w0, convt0, D, D);
    k_transpose<<<dim3(D / 16, D / 16, 5), tb16, 0, stream>>>(conv_w1, convt1, D, D);

    const int gcv = (int)((M * D + 255) / 256);
    const int gcs = (int)((M * DC + 255) / 256);
    const dim3 ggD(D / 128, (unsigned)(M / 128));   // GEMM grid, N=D
    const dim3 ggC(DC / 128, (unsigned)(M / 128));  // GEMM grid, N=DC

    // ---- LTM read path ----
    k_f2bf<<<gcv, 256, 0, stream>>>(x, ab0, M * D);
    k_gemm<<<ggC, 256, 0, stream>>>(ab0, wq_ltm_t, f2, D, DC);          // q
    k_rowgate<<<(unsigned)M, 256, 0, stream>>>(x, wgr_ltm, gA, D);
    k_attn_read<<<(unsigned)M, 64, 0, stream>>>(f2, cache, f3, sc);     // read
    k_f2bf<<<gcs, 256, 0, stream>>>(f3, ab1, M * DC);
    k_gemm<<<ggD, 256, 0, stream>>>(ab1, wo_ltm_t, f0, DC, D);          // read@Wo
    k_addgate<<<gcv, 256, 0, stream>>>(x, gA, f0, f1, M * D, D);        // x1

    // ---- WM read path ----
    k_f2bf<<<gcv, 256, 0, stream>>>(f1, ab0, M * D);
    k_gemm<<<ggC, 256, 0, stream>>>(ab0, wq_wm_t, f2, D, DC);           // q2
    k_rowgate<<<(unsigned)M, 256, 0, stream>>>(f1, wgr_wm, gB, D);
    k_attn_wm<<<(unsigned)M, 64, 0, stream>>>(f2, wm, f3);              // read2
    k_f2bf<<<gcs, 256, 0, stream>>>(f3, ab1, M * DC);
    k_gemm<<<ggD, 256, 0, stream>>>(ab1, wo_wm_t, f0, DC, D);           // read2@Wo
    k_addgate<<<gcv, 256, 0, stream>>>(f1, gB, f0, f1, M * D, D);       // x2 (in place)

    // ---- dilated conv stack (WMMA-dominant) ----
    k_ln<<<(unsigned)M, 256, 0, stream>>>(f1, ln_s0, ln_b0, ab0, nullptr);
    k_conv<<<ggD, 256, 0, stream>>>(ab0, convt0, conv_b0, f1, 1);
    k_ln<<<(unsigned)M, 256, 0, stream>>>(f1, ln_s1, ln_b1, ab0, nullptr);
    k_conv<<<ggD, 256, 0, stream>>>(ab0, convt1, conv_b1, f1, 2);

    // ---- post LN -> out ----
    k_ln<<<(unsigned)M, 256, 0, stream>>>(f1, post_s, post_b, ab0, out);

    // ---- write paths ----
    k_gemm<<<ggC, 256, 0, stream>>>(ab0, wv_wm_t, f2, D, DC);           // vw
    k_gemm<<<ggC, 256, 0, stream>>>(ab0, wv_ltm_t, f3, D, DC);          // vl
    k_rowgate<<<(unsigned)M, 256, 0, stream>>>(out, wgw_wm, gA, D);
    k_rowgate<<<(unsigned)M, 256, 0, stream>>>(out, wgw_ltm, gB, D);
    k_wv<<<(unsigned)M, 64, 0, stream>>>(out, Ws_wm, gA, wvb);          // wv
    k_attn_w<<<(unsigned)M, 64, 0, stream>>>(f3, cache, gB, wlb, sc);   // wl
    k_upd_wm<<<Bb * NK, 256, 0, stream>>>(wvb, f2, wm, owm);
    k_upd_cache<<<Bb * KK, 256, 0, stream>>>(wlb, f3, cache, ocache);
}
